// MultiHeadAttention_43293270343852
// MI455X (gfx1250) — matrix operand order repaired and checked by reading, not compiled, not run on hardware
//
#include <hip/hip_runtime.h>
#include <hip/hip_bf16.h>

// ---------------------------------------------------------------------------
// MHA (B=16, S=1024, D=512, H=8, DH=64) on gfx1250 using bf16 WMMA.
//   pass 1: Q/K/V projections + LeakyReLU, f32 -> bf16 in d_ws
//           (Q,K: [B,H,S,DH];  V: transposed [B,H,DH,S] so attention can
//            stage it with contiguous async global->LDS copies)
//   pass 2: flash attention (online softmax with DPP16 lane reductions),
//           double-buffered K/V tiles staged with
//           GLOBAL_LOAD_ASYNC_TO_LDS_B128 (ASYNCcnt path), f32 out [B,S,D]
// ---------------------------------------------------------------------------

#define BB 16
#define SS 1024
#define DD 512
#define HH 8
#define DHH 64
#define NEGV_SCALED (-536870911.875f)  // (-2^32+1) * 0.125, mask folded in FMA
#define LOG2E 1.44269504088896f

typedef __attribute__((ext_vector_type(16))) __bf16 v16bf;
typedef __attribute__((ext_vector_type(8)))  __bf16 v8bf;
typedef __attribute__((ext_vector_type(8)))  float  v8f;

__device__ __forceinline__ v16bf combine16(v8bf lo, v8bf hi) {
    return __builtin_shufflevector(lo, hi, 0,1,2,3,4,5,6,7,8,9,10,11,12,13,14,15);
}

// A-fragment (16x32 bf16): lanes 0-15 hold K 0..7 / 16..23, lanes 16-31 hold
// K 8..15 / 24..31, row m = lane&15. rowp -> (row m, k=0), 16B aligned.
__device__ __forceinline__ v16bf load_a_frag(const __bf16* rowp, int half) {
    const v8bf* p0 = (const v8bf*)(rowp + half * 8);
    const v8bf* p1 = (const v8bf*)(rowp + 16 + half * 8);
    return combine16(p0[0], p1[0]);
}

// B-fragment (32x16 bf16) from Bt[n][k] (n-major, contiguous K).
// Lane holds column n = lane&15; lanes 0-15 K 0..7 / 16..23, lanes 16-31 K 8..15 / 24..31.
__device__ __forceinline__ v16bf load_b_frag(const __bf16* rowp, int half) {
    const v8bf* p0 = (const v8bf*)(rowp + half * 8);
    const v8bf* p1 = (const v8bf*)(rowp + 16 + half * 8);
    return combine16(p0[0], p1[0]);
}

// ---- DPP16 16-lane butterfly reductions (no LDS, pure VALU) ---------------
// quad_perm(1,0,3,2)=0xB1 (xor1), quad_perm(2,3,0,1)=0x4E (xor2),
// row_half_mirror=0x141 (eff. xor4), row_mirror=0x140 (eff. xor8).
template <int CTRL>
__device__ __forceinline__ float dpp_movf(float v) {
    int y = __builtin_amdgcn_update_dpp(0, __builtin_bit_cast(int, v),
                                        CTRL, 0xF, 0xF, true);
    return __builtin_bit_cast(float, y);
}
__device__ __forceinline__ float red_max16(float v) {
    v = fmaxf(v, dpp_movf<0xB1>(v));
    v = fmaxf(v, dpp_movf<0x4E>(v));
    v = fmaxf(v, dpp_movf<0x141>(v));
    v = fmaxf(v, dpp_movf<0x140>(v));
    return v;
}
__device__ __forceinline__ float red_sum16(float v) {
    v += dpp_movf<0xB1>(v);
    v += dpp_movf<0x4E>(v);
    v += dpp_movf<0x141>(v);
    v += dpp_movf<0x140>(v);
    return v;
}

// ---- CDNA5 async global->LDS copy (16B per lane, per-lane addresses) ------
__device__ __forceinline__ void async_copy_b128(uint32_t lds_off, const void* gptr) {
    asm volatile("global_load_async_to_lds_b128 %0, %1, off"
                 :: "v"(lds_off), "v"((unsigned long long)(uintptr_t)gptr)
                 : "memory");
}
__device__ __forceinline__ void wait_async_le4() {
    asm volatile("s_wait_asynccnt 0x4" ::: "memory");
}
__device__ __forceinline__ void wait_async_0() {
    asm volatile("s_wait_asynccnt 0x0" ::: "memory");
}
__device__ __forceinline__ uint32_t lds_offset(const void* p) {
    return (uint32_t)(uintptr_t)p;  // generic LDS addr: low 32 bits = LDS offset
}

// ---------------------------------------------------------------------------
// Projection: leaky( X[r,:] . W[o,:] + bias[o] ) -> bf16.
// Block tile 128 rows x 64 cols, 8 waves; each wave: 1 row-subtile x 4 col
// subtiles => 4 WMMA per K-step with the A-fragment reused 4x.
// VT=false: dst[b,h,s,dh].  VT=true: dst[b,h,dh,s] (transposed for V).
// ---------------------------------------------------------------------------
#define PSTR 56   // LDS row stride (elems): 112B, 28 words -> conflict-free
template <bool VT>
__global__ __launch_bounds__(256) void proj_leaky_kernel(
    const float* __restrict__ X, const float* __restrict__ W,
    const float* __restrict__ bias, __bf16* __restrict__ dst)
{
    __shared__ __align__(32) __bf16 As[128][PSTR];  // [row][k]
    __shared__ __align__(32) __bf16 Bs[64][PSTR];   // [col][k] == rows of W

    const int r0   = blockIdx.x * 128;
    const int o0   = blockIdx.y * 64;
    const int tid  = threadIdx.x;
    const int lane = tid & 31;
    const int wv   = tid >> 5;
    const int half = lane >> 4;
    const int l16  = lane & 15;

    v8f acc[4] = {};

    const int arow = tid >> 1;          // 0..127
    const int aseg = (tid & 1) * 16;    // 0,16
    const int brow = tid >> 2;          // 0..63
    const int bseg = (tid & 3) * 8;     // 0,8,16,24

    for (int k0 = 0; k0 < DD; k0 += 32) {
        {   // stage + convert f32 -> bf16
            const float* xs = X + (size_t)(r0 + arow) * DD + k0 + aseg;
            v8bf xa, xb;
#pragma unroll
            for (int j = 0; j < 8; ++j) { xa[j] = (__bf16)xs[j]; xb[j] = (__bf16)xs[8 + j]; }
            *(v8bf*)&As[arow][aseg + 0] = xa;
            *(v8bf*)&As[arow][aseg + 8] = xb;
            const float* ws = W + (size_t)(o0 + brow) * DD + k0 + bseg;
            v8bf wa;
#pragma unroll
            for (int j = 0; j < 8; ++j) wa[j] = (__bf16)ws[j];
            *(v8bf*)&Bs[brow][bseg] = wa;
        }
        __syncthreads();

        v16bf a = load_a_frag(&As[wv * 16 + l16][0], half);
#pragma unroll
        for (int t = 0; t < 4; ++t) {
            v16bf b = load_b_frag(&Bs[t * 16 + l16][0], half);
            acc[t] = __builtin_amdgcn_wmma_f32_16x16x32_bf16(false, a, false, b,
                                                             (short)0, acc[t], false, false);
        }
        __syncthreads();
    }

    // epilogue: bias + LeakyReLU(0.2)
#pragma unroll
    for (int t = 0; t < 4; ++t) {
        const int col = o0 + t * 16 + l16;
        const float bv = bias[col];
        const int h  = col >> 6;
        const int dh = col & (DHH - 1);
#pragma unroll
        for (int i = 0; i < 8; ++i) {
            const int r = r0 + wv * 16 + half * 8 + i;
            const int b = r >> 10;
            const int s = r & (SS - 1);
            float v = acc[t][i] + bv;
            v = (v >= 0.0f) ? v : 0.2f * v;
            const size_t idx = VT
                ? ((size_t)((b * HH + h) * DHH + dh)) * SS + s
                : ((size_t)((b * HH + h) * SS + s)) * DHH + dh;
            dst[idx] = (__bf16)v;
        }
    }
}

// ---------------------------------------------------------------------------
// Flash attention. Block = 128 q rows of one (b,h); 8 waves x 16 rows.
// K-tile = 64 keys, double-buffered in LDS via async global->LDS copies.
// scores=(Q.K^T + mask)/8, online softmax, O += P.V.
// ---------------------------------------------------------------------------
#define ASTR 72   // LDS row stride (elems): 144B, 36 words -> conflict-free
__global__ __launch_bounds__(256) void flash_attn_kernel(
    const __bf16* __restrict__ qh, const __bf16* __restrict__ kh,
    const __bf16* __restrict__ vht, const float* __restrict__ qmask,
    const float* __restrict__ kmask, float* __restrict__ out)
{
    __shared__ __align__(32) __bf16 Ks[2][64][ASTR];    // [buf][key][dh]
    __shared__ __align__(32) __bf16 Vt[2][DHH][ASTR];   // [buf][dh][key]
    __shared__ __align__(32) __bf16 Ps[8][16][ASTR];    // per-wave P tile

    const int b  = blockIdx.z;
    const int h  = blockIdx.y;
    const int q0 = blockIdx.x * 128;
    const int tid  = threadIdx.x;
    const int lane = tid & 31;
    const int wv   = tid >> 5;
    const int half = lane >> 4;
    const int l16  = lane & 15;
    const int qrow0 = q0 + wv * 16;

    const size_t headOff = (size_t)((b * HH + h) * SS) * DHH;
    const __bf16* qbase = qh + headOff;
    const __bf16* kbase = kh + headOff;
    const __bf16* vbase = vht + headOff;  // [dh][s] rows of length S

    // staging chunk map: 512 x 16B chunks per tile, 2 per thread
    const int c0 = tid * 2;

    // Q fragments: 2 A-frags covering dh 0..63 for this wave's 16 rows
    v16bf qa[2];
#pragma unroll
    for (int s = 0; s < 2; ++s)
        qa[s] = load_a_frag(qbase + (size_t)(qrow0 + l16) * DHH + s * 32, half);

    float qm[8];
#pragma unroll
    for (int i = 0; i < 8; ++i) qm[i] = qmask[b * SS + qrow0 + half * 8 + i];

    float mrow[8], lrow[8];
#pragma unroll
    for (int i = 0; i < 8; ++i) { mrow[i] = -3.0e38f; lrow[i] = 0.0f; }
    v8f oacc[4] = {};

    // ---- issue async prefetch of tile 0 into buffer 0
#pragma unroll
    for (int j = 0; j < 2; ++j) {
        const int row = ((c0 + j) >> 3), seg = ((c0 + j) & 7) * 8;
        async_copy_b128(lds_offset(&Ks[0][row][seg]),
                        kbase + (size_t)row * DHH + seg);
    }
#pragma unroll
    for (int j = 0; j < 2; ++j) {
        const int row = ((c0 + j) >> 3), seg = ((c0 + j) & 7) * 8;
        async_copy_b128(lds_offset(&Vt[0][row][seg]),
                        vbase + (size_t)row * SS + seg);
    }

    const int NT = SS / 64;
    for (int t = 0; t < NT; ++t) {
        const int buf = t & 1;
        const int kt  = t * 64;

        if (t + 1 < NT) {  // prefetch next tile into the other buffer
            const int nb = buf ^ 1;
            const int nkt = kt + 64;
#pragma unroll
            for (int j = 0; j < 2; ++j) {
                const int row = ((c0 + j) >> 3), seg = ((c0 + j) & 7) * 8;
                async_copy_b128(lds_offset(&Ks[nb][row][seg]),
                                kbase + (size_t)(nkt + row) * DHH + seg);
            }
#pragma unroll
            for (int j = 0; j < 2; ++j) {
                const int row = ((c0 + j) >> 3), seg = ((c0 + j) & 7) * 8;
                async_copy_b128(lds_offset(&Vt[nb][row][seg]),
                                vbase + (size_t)row * SS + nkt + seg);
            }
            wait_async_le4();   // in-order async => current tile landed
        } else {
            wait_async_0();
        }
        __syncthreads();

        // ---- scores: 16 q rows x 64 keys = 4 C-frags, K-dim = 2 steps
        v8f sc[4] = {};
#pragma unroll
        for (int n = 0; n < 4; ++n)
#pragma unroll
            for (int s = 0; s < 2; ++s) {
                v16bf bk = load_b_frag(&Ks[buf][n * 16 + l16][s * 32], half);
                sc[n] = __builtin_amdgcn_wmma_f32_16x16x32_bf16(
                    false, qa[s], false, bk, (short)0, sc[n], false, false);
            }

        // ---- mask (added BEFORE scaling, folded into FMA) + online softmax
        float km[4];
#pragma unroll
        for (int n = 0; n < 4; ++n) km[n] = kmask[b * SS + kt + n * 16 + l16];

        float alpha[8];
#pragma unroll
        for (int i = 0; i < 8; ++i) {
            float mloc = -3.0e38f;
#pragma unroll
            for (int n = 0; n < 4; ++n) {
                const float addm = (qm[i] * km[n] != 0.0f) ? 0.0f : NEGV_SCALED;
                const float sv = fmaf(sc[n][i], 0.125f, addm);
                sc[n][i] = sv;
                mloc = fmaxf(mloc, sv);
            }
            mloc = red_max16(mloc);                       // DPP16, no LDS
            const float mnew = fmaxf(mrow[i], mloc);
            const float a = exp2f((mrow[i] - mnew) * LOG2E);
            alpha[i] = a;
            float rsum = 0.0f;
#pragma unroll
            for (int n = 0; n < 4; ++n) {
                const float p = exp2f((sc[n][i] - mnew) * LOG2E);
                sc[n][i] = p;
                rsum += p;
            }
            rsum = red_sum16(rsum);                       // DPP16, no LDS
            lrow[i] = lrow[i] * a + rsum;
            mrow[i] = mnew;
        }
#pragma unroll
        for (int d = 0; d < 4; ++d)
#pragma unroll
            for (int i = 0; i < 8; ++i) oacc[d][i] = oacc[d][i] * alpha[i];

        // ---- P (C layout) -> LDS -> A layout (same-wave DS ops in order)
#pragma unroll
        for (int n = 0; n < 4; ++n)
#pragma unroll
            for (int i = 0; i < 8; ++i)
                Ps[wv][half * 8 + i][n * 16 + l16] = (__bf16)sc[n][i];

        // ---- O += P @ V
#pragma unroll
        for (int s = 0; s < 2; ++s) {
            v16bf pa = load_a_frag(&Ps[wv][l16][s * 32], half);
#pragma unroll
            for (int d = 0; d < 4; ++d) {
                v16bf bv = load_b_frag(&Vt[buf][d * 16 + l16][s * 32], half);
                oacc[d] = __builtin_amdgcn_wmma_f32_16x16x32_bf16(
                    false, pa, false, bv, (short)0, oacc[d], false, false);
            }
        }
        __syncthreads();  // reads of buf done before its next prefetch
    }

    // epilogue: /l, * q_mask (post-softmax), write f32 [B,S,D]
    float* obase = out + (size_t)(b * SS) * DD + (size_t)h * DHH;
#pragma unroll
    for (int i = 0; i < 8; ++i) {
        const int r = qrow0 + half * 8 + i;
        const float inv = qm[i] / fmaxf(lrow[i], 1e-30f);
#pragma unroll
        for (int d = 0; d < 4; ++d)
            obase[(size_t)r * DD + d * 16 + l16] = oacc[d][i] * inv;
    }
}

// ---------------------------------------------------------------------------
extern "C" void kernel_launch(void* const* d_in, const int* in_sizes, int n_in,
                              void* d_out, int out_size, void* d_ws, size_t ws_size,
                              hipStream_t stream) {
    const float* query  = (const float*)d_in[0];
    const float* key    = (const float*)d_in[1];
    const float* value  = (const float*)d_in[2];
    const float* q_mask = (const float*)d_in[3];
    const float* k_mask = (const float*)d_in[4];
    const float* WQ = (const float*)d_in[5];
    const float* bQ = (const float*)d_in[6];
    const float* WK = (const float*)d_in[7];
    const float* bK = (const float*)d_in[8];
    const float* WV = (const float*)d_in[9];
    const float* bV = (const float*)d_in[10];
    float* out = (float*)d_out;

    const size_t headElems = (size_t)BB * HH * SS * DHH;
    __bf16* qhw = (__bf16*)d_ws;
    __bf16* khw = qhw + headElems;
    __bf16* vhw = khw + headElems;   // transposed [B,H,DH,S]

    dim3 pgrid((BB * SS) / 128, DD / 64);   // 128 x 8
    proj_leaky_kernel<false><<<pgrid, 256, 0, stream>>>(query, WQ, bQ, qhw);
    proj_leaky_kernel<false><<<pgrid, 256, 0, stream>>>(key,   WK, bK, khw);
    proj_leaky_kernel<true ><<<pgrid, 256, 0, stream>>>(value, WV, bV, vhw);

    dim3 agrid(SS / 128, HH, BB);           // 8 x 8 x 16
    flash_attn_kernel<<<agrid, 256, 0, stream>>>(qhw, khw, vhw, q_mask, k_mask, out);
}
